// MoELoRALinear_86234353369531
// MI455X (gfx1250) — matrix operand order repaired and checked by reading, not compiled, not run on hardware
//
#include <hip/hip_runtime.h>
#include <hip/hip_bf16.h>
#include <stdint.h>

#define USE_ASYNC 1   // gfx1250 GLOBAL_LOAD_ASYNC_TO_LDS_B128 path; set 0 to fall back

typedef unsigned int uint32;
typedef __attribute__((ext_vector_type(16))) __bf16 v16bf;
typedef __attribute__((ext_vector_type(8)))  float  v8f;

// Problem constants (match reference)
constexpr int B_ = 4, T_ = 2048, IN_ = 4096, OUT_ = 4096, E_ = 8, R_ = 16;
constexpr int MTOT = B_ * T_;          // 8192 tokens
constexpr int NCAT = 144;              // 128 lora-A rows + 8 router rows + 8 zero pad
constexpr int LPITCH = 40;             // LDS row pitch in halfs (80B) -> conflict-free b128 reads
constexpr int NKMAIN = IN_ / 32;       // 128
constexpr int NKTOT  = NKMAIN + 4;     // + 4 LoRA K-steps (K=128)

// ---- fp32 -> bf16 round-to-nearest-even ----
__device__ __forceinline__ unsigned short f2bf(float f) {
  uint32 u = __builtin_bit_cast(uint32, f);
  u += 0x7FFFu + ((u >> 16) & 1u);
  return (unsigned short)(u >> 16);
}
__device__ __forceinline__ uint32 pk2(float lo, float hi) {
  return (uint32)f2bf(lo) | ((uint32)f2bf(hi) << 16);
}

union FragU { uint4 u[2]; v16bf v; };

// ---- async global->LDS copy (CDNA5 ASYNCcnt path) ----
__device__ __forceinline__ void async_b128(void* lds, const void* g) {
#if USE_ASYNC
  unsigned lofs = (unsigned)(unsigned long long)lds;   // low 32 bits of generic = LDS offset
  unsigned long long ga = (unsigned long long)g;
  asm volatile("global_load_async_to_lds_b128 %0, %1, off" :: "v"(lofs), "v"(ga) : "memory");
#else
  *(uint4*)lds = *(const uint4*)g;
#endif
}
__device__ __forceinline__ void async_wait0() {
#if USE_ASYNC
  asm volatile("s_wait_asynccnt 0" ::: "memory");
#endif
}

// ---------------- pack / convert kernels ----------------
__global__ void pack_wcat(const float* __restrict__ A_w, const float* __restrict__ W_router,
                          unsigned short* __restrict__ Wcat) {
  int idx = blockIdx.x * blockDim.x + threadIdx.x;
  if (idx >= NCAT * IN_) return;
  int row = idx / IN_, col = idx - row * IN_;
  float v = 0.f;
  if (row < 128)      v = A_w[(size_t)row * IN_ + col];           // A_w flat [E*R=128, IN]
  else if (row < 136) v = W_router[(size_t)(row - 128) * IN_ + col];
  Wcat[idx] = f2bf(v);
}

__global__ void pack_bcat(const float* __restrict__ B_w, unsigned short* __restrict__ Bcat) {
  int idx = blockIdx.x * blockDim.x + threadIdx.x;
  if (idx >= OUT_ * 128) return;
  int o = idx >> 7, c = idx & 127;
  int e = c >> 4, r = c & 15;
  Bcat[idx] = f2bf(B_w[((size_t)e * OUT_ + o) * R_ + r]);         // B_w [E, OUT, R]
}

// convert n8*8 fp32 -> bf16 (vectorized: 2x float4 in, 1x uint4 out per thread)
__global__ void conv_bf16(const float* __restrict__ src, unsigned short* __restrict__ dst, int n8) {
  int i = blockIdx.x * blockDim.x + threadIdx.x;
  if (i >= n8) return;
  const float4* s = (const float4*)src + (size_t)i * 2;
  float4 a = s[0], b = s[1];
  uint4 o;
  o.x = pk2(a.x, a.y); o.y = pk2(a.z, a.w);
  o.z = pk2(b.x, b.y); o.w = pk2(b.z, b.w);
  ((uint4*)dst)[i] = o;
}

// ------- GEMM A: P[8192,144] = x @ Wcat^T (bf16 WMMA, f32 accum) -------
template<bool PRE>
__global__ __launch_bounds__(128)
void lora_router_gemm(const float* __restrict__ x, const unsigned short* __restrict__ xb,
                      const unsigned short* __restrict__ Wcat, float* __restrict__ P) {
  __shared__ __align__(16) unsigned short As[64 * LPITCH];
  __shared__ __align__(16) unsigned short Bs[144 * LPITCH];
  const int tid = threadIdx.x, lane = tid & 31, wid = tid >> 5;
  const int m0 = blockIdx.x * 64;

  v8f acc[9] = {};

  for (int kt = 0; kt < NKMAIN; ++kt) {
    const int k0 = kt * 32;
    __syncthreads();
    if constexpr (PRE) {
      // A tile from bf16 x: 256 8-half chunks / 128 thr = 2
#pragma unroll
      for (int i = 0; i < 2; ++i) {
        int c = tid + i * 128;
        int row = c >> 2, cc = c & 3;
        *(uint4*)&As[row * LPITCH + cc * 8] =
            *(const uint4*)&xb[(size_t)(m0 + row) * IN_ + k0 + cc * 8];
      }
    } else {
      // A tile from fp32 x with on-the-fly conversion: 512 float4-chunks / 128 thr = 4
#pragma unroll
      for (int i = 0; i < 4; ++i) {
        int c = tid + i * 128;
        int row = c >> 3, f4 = c & 7;
        float4 a = *(const float4*)&x[(size_t)(m0 + row) * IN_ + k0 + f4 * 4];
        uint32* pa = (uint32*)&As[row * LPITCH + f4 * 4];
        pa[0] = pk2(a.x, a.y);
        pa[1] = pk2(a.z, a.w);
      }
    }
    // B tile: 144 rows x 32 bf16. 576 8-half chunks.
    for (int c = tid; c < 576; c += 128) {
      int row = c >> 2, cc = c & 3;
      *(uint4*)&Bs[row * LPITCH + cc * 8] =
          *(const uint4*)&Wcat[(size_t)row * IN_ + k0 + cc * 8];
    }
    __syncthreads();

    const int l15 = lane & 15;
    const int aoff = (lane >= 16) ? 8 : 0;
    const int boff = (lane >= 16) ? 8 : 0;
    FragU af;
    const int arow = wid * 16 + l15;
    af.u[0] = *(const uint4*)&As[arow * LPITCH + aoff];
    af.u[1] = *(const uint4*)&As[arow * LPITCH + aoff + 16];
#pragma unroll
    for (int n = 0; n < 9; ++n) {
      FragU bf;
      const int brow = n * 16 + l15;
      bf.u[0] = *(const uint4*)&Bs[brow * LPITCH + boff];
      bf.u[1] = *(const uint4*)&Bs[brow * LPITCH + boff + 16];
      acc[n] = __builtin_amdgcn_wmma_f32_16x16x32_bf16(
          false, af.v, false, bf.v, (short)0, acc[n], false, false);
    }
  }
  const int col = lane & 15;
  const int rbase = m0 + wid * 16 + ((lane >= 16) ? 8 : 0);
#pragma unroll
  for (int n = 0; n < 9; ++n)
#pragma unroll
    for (int j = 0; j < 8; ++j)
      P[(size_t)(rbase + j) * NCAT + n * 16 + col] = acc[n][j];
}

// ------- gating: top-2 softmax over logits (P cols 128..135), scale P -> si bf16 -------
__global__ void gate_kernel(const float* __restrict__ P, unsigned short* __restrict__ si) {
  int t = blockIdx.x * blockDim.x + threadIdx.x;
  if (t >= MTOT) return;
  const float* L = &P[(size_t)t * NCAT + 128];
  float v0 = -3.0e38f; int i0 = 0;
#pragma unroll
  for (int e = 0; e < 8; ++e) { float l = L[e]; if (l > v0) { v0 = l; i0 = e; } }
  float v1 = -3.0e38f; int i1 = 0;
#pragma unroll
  for (int e = 0; e < 8; ++e) { if (e == i0) continue; float l = L[e]; if (l > v1) { v1 = l; i1 = e; } }
  float e1 = __expf(v1 - v0);
  float inv = 1.f / (1.f + e1);
  float w0 = inv;            // SCALING = alpha/r = 1.0
  float w1 = e1 * inv;
  const float* Pi = &P[(size_t)t * NCAT];
  unsigned short* s = &si[(size_t)t * 128];
#pragma unroll
  for (int c = 0; c < 128; ++c) {
    int e = c >> 4;
    float w = (e == i0) ? w0 : (e == i1) ? w1 : 0.f;
    s[c] = f2bf(w * Pi[c]);
  }
}

// ------- shared WMMA micro-kernel: 8 tiles per wave on a 128x128x32 block tile -------
__device__ __forceinline__ void compute_step(const unsigned short* As, const unsigned short* Bs,
                                             v8f acc[4][2], int lane, int wm, int wn) {
  const int l15 = lane & 15;
  const int aoff = (lane >= 16) ? 8 : 0;
  const int boff = (lane >= 16) ? 8 : 0;
  FragU af[4], bf[2];
#pragma unroll
  for (int mt = 0; mt < 4; ++mt) {
    const int arow = wm * 64 + mt * 16 + l15;
    af[mt].u[0] = *(const uint4*)&As[arow * LPITCH + aoff];
    af[mt].u[1] = *(const uint4*)&As[arow * LPITCH + aoff + 16];
  }
#pragma unroll
  for (int nt = 0; nt < 2; ++nt) {
    const int brow = wn * 32 + nt * 16 + l15;
    bf[nt].u[0] = *(const uint4*)&Bs[brow * LPITCH + boff];
    bf[nt].u[1] = *(const uint4*)&Bs[brow * LPITCH + boff + 16];
  }
#pragma unroll
  for (int mt = 0; mt < 4; ++mt)
#pragma unroll
    for (int nt = 0; nt < 2; ++nt)
      acc[mt][nt] = __builtin_amdgcn_wmma_f32_16x16x32_bf16(
          false, af[mt].v, false, bf[nt].v, (short)0, acc[mt][nt], false, false);
}

// ------- main fused GEMM: out = x @ W_base^T + b_base + si @ Bcat^T -------
template<bool PRE>
__global__ __launch_bounds__(256)
void main_gemm(const float* __restrict__ x, const float* __restrict__ W_base,
               const unsigned short* __restrict__ xb, const unsigned short* __restrict__ Wb,
               const float* __restrict__ b_base, const unsigned short* __restrict__ si,
               const unsigned short* __restrict__ Bcat, float* __restrict__ out) {
  __shared__ __align__(16) unsigned short As[2][128 * LPITCH];
  __shared__ __align__(16) unsigned short Bs[2][128 * LPITCH];
  const int tid = threadIdx.x, lane = tid & 31, wid = tid >> 5;
  const int wm = wid & 1, wn = wid >> 1;          // 2 x 4 wave grid
  const int m0 = blockIdx.x * 128, n0 = blockIdx.y * 128;

  v8f acc[4][2] = {};

  if constexpr (PRE) {
    // --- all-bf16 inputs, double-buffered, async global->LDS pipeline ---
    auto issue = [&](int kt, int p) {
#pragma unroll
      for (int i = 0; i < 2; ++i) {               // 512 16B chunks per matrix / 256 thr
        int c = tid + i * 256;
        int row = c >> 2, cc = c & 3;
        const unsigned short *ga, *gb;
        if (kt < NKMAIN) {
          ga = xb + (size_t)(m0 + row) * IN_ + kt * 32 + cc * 8;
          gb = Wb + (size_t)(n0 + row) * IN_ + kt * 32 + cc * 8;
        } else {
          const int k0 = (kt - NKMAIN) * 32;
          ga = si   + (size_t)(m0 + row) * 128 + k0 + cc * 8;
          gb = Bcat + (size_t)(n0 + row) * 128 + k0 + cc * 8;
        }
        async_b128(&As[p][row * LPITCH + cc * 8], ga);
        async_b128(&Bs[p][row * LPITCH + cc * 8], gb);
      }
    };
    issue(0, 0);
    for (int kt = 0; kt < NKTOT; ++kt) {
      const int p = kt & 1;
      async_wait0();             // my tile-kt copies done
      __syncthreads();           // everyone's copies done; everyone done with buf p^1
      if (kt + 1 < NKTOT) issue(kt + 1, p ^ 1);
      compute_step(As[p], Bs[p], acc, lane, wm, wn);
#if !USE_ASYNC
      __syncthreads();           // fallback copies are plain ds stores: keep them race-free
#endif
    }
  } else {
    // --- fallback: fp32 sources, convert in-loop (single buffer, 2 barriers/iter) ---
    for (int kt = 0; kt < NKMAIN; ++kt) {
      const int k0 = kt * 32;
      float4 a[4], b[4];
#pragma unroll
      for (int i = 0; i < 4; ++i) {
        int c = tid + i * 256;
        int row = c >> 3, f4 = c & 7;
        a[i] = *(const float4*)&x[(size_t)(m0 + row) * IN_ + k0 + f4 * 4];
        b[i] = *(const float4*)&W_base[(size_t)(n0 + row) * IN_ + k0 + f4 * 4];
      }
      __syncthreads();
#pragma unroll
      for (int i = 0; i < 4; ++i) {
        int c = tid + i * 256;
        int row = c >> 3, f4 = c & 7;
        uint32* pa = (uint32*)&As[0][row * LPITCH + f4 * 4];
        pa[0] = pk2(a[i].x, a[i].y); pa[1] = pk2(a[i].z, a[i].w);
        uint32* pb = (uint32*)&Bs[0][row * LPITCH + f4 * 4];
        pb[0] = pk2(b[i].x, b[i].y); pb[1] = pk2(b[i].z, b[i].w);
      }
      __syncthreads();
      compute_step(As[0], Bs[0], acc, lane, wm, wn);
    }
    for (int kt = 0; kt < 4; ++kt) {
      const int k0 = kt * 32;
      uint4 a[2], b[2];
#pragma unroll
      for (int i = 0; i < 2; ++i) {
        int c = tid + i * 256;
        int row = c >> 2, cc = c & 3;
        a[i] = *(const uint4*)&si[(size_t)(m0 + row) * 128 + k0 + cc * 8];
        b[i] = *(const uint4*)&Bcat[(size_t)(n0 + row) * 128 + k0 + cc * 8];
      }
      __syncthreads();
#pragma unroll
      for (int i = 0; i < 2; ++i) {
        int c = tid + i * 256;
        int row = c >> 2, cc = c & 3;
        *(uint4*)&As[0][row * LPITCH + cc * 8] = a[i];
        *(uint4*)&Bs[0][row * LPITCH + cc * 8] = b[i];
      }
      __syncthreads();
      compute_step(As[0], Bs[0], acc, lane, wm, wn);
    }
  }

  // ---- epilogue: + b_base, write fp32 ----
  const int col = lane & 15;
  const int hi = (lane >= 16) ? 8 : 0;
#pragma unroll
  for (int mt = 0; mt < 4; ++mt) {
    const int row0 = m0 + wm * 64 + mt * 16 + hi;
#pragma unroll
    for (int nt = 0; nt < 2; ++nt) {
      const int c0 = n0 + wn * 32 + nt * 16 + col;
      const float bb = b_base[c0];
#pragma unroll
      for (int j = 0; j < 8; ++j)
        out[(size_t)(row0 + j) * OUT_ + c0] = acc[mt][nt][j] + bb;
    }
  }
}

// ---------------- launcher ----------------
extern "C" void kernel_launch(void* const* d_in, const int* in_sizes, int n_in,
                              void* d_out, int out_size, void* d_ws, size_t ws_size,
                              hipStream_t stream) {
  const float* x        = (const float*)d_in[0];
  const float* W_base   = (const float*)d_in[1];
  const float* b_base   = (const float*)d_in[2];
  const float* W_router = (const float*)d_in[3];
  const float* A_w      = (const float*)d_in[4];
  const float* B_w      = (const float*)d_in[5];
  float* out = (float*)d_out;

  // workspace layout
  char* ws = (char*)d_ws;
  const size_t szP    = (size_t)MTOT * NCAT * 4;   // 4.7 MB
  const size_t szSi   = (size_t)MTOT * 128 * 2;    // 2 MB
  const size_t szWcat = (size_t)NCAT * IN_ * 2;    // 1.2 MB
  const size_t szBcat = (size_t)OUT_ * 128 * 2;    // 1 MB
  const size_t szXb   = (size_t)MTOT * IN_ * 2;    // 64 MB
  const size_t szWb   = (size_t)OUT_ * IN_ * 2;    // 32 MB
  float*          P    = (float*)ws;
  unsigned short* si   = (unsigned short*)(ws + szP);
  unsigned short* Wcat = (unsigned short*)(ws + szP + szSi);
  unsigned short* Bcat = (unsigned short*)(ws + szP + szSi + szWcat);
  unsigned short* xbuf = (unsigned short*)(ws + szP + szSi + szWcat + szBcat);
  unsigned short* Wb   = (unsigned short*)(ws + szP + szSi + szWcat + szBcat + szXb);
  const bool pre = ws_size >= szP + szSi + szWcat + szBcat + szXb + szWb;

  pack_wcat<<<(NCAT * IN_ + 255) / 256, 256, 0, stream>>>(A_w, W_router, Wcat);
  pack_bcat<<<(OUT_ * 128 + 255) / 256, 256, 0, stream>>>(B_w, Bcat);

  dim3 grid(MTOT / 128, OUT_ / 128);
  if (pre) {
    conv_bf16<<<(MTOT * (IN_ / 8) + 255) / 256, 256, 0, stream>>>(x, xbuf, MTOT * IN_ / 8);
    conv_bf16<<<(OUT_ * (IN_ / 8) + 255) / 256, 256, 0, stream>>>(W_base, Wb, OUT_ * IN_ / 8);
    lora_router_gemm<true><<<MTOT / 64, 128, 0, stream>>>(x, xbuf, Wcat, P);
    gate_kernel<<<MTOT / 256, 256, 0, stream>>>(P, si);
    main_gemm<true><<<grid, 256, 0, stream>>>(x, W_base, xbuf, Wb, b_base, si, Bcat, out);
  } else {
    lora_router_gemm<false><<<MTOT / 64, 128, 0, stream>>>(x, xbuf, Wcat, P);
    gate_kernel<<<MTOT / 256, 256, 0, stream>>>(P, si);
    main_gemm<false><<<grid, 256, 0, stream>>>(x, W_base, xbuf, Wb, b_base, si, Bcat, out);
  }
}
